// PortHamiltonianModel_39900246180480
// MI455X (gfx1250) — matrix operand order repaired and checked by reading, not compiled, not run on hardware
//
#include <hip/hip_runtime.h>
#include <hip/hip_bf16.h>

#define BATCH   131072
#define HDIM    512
#define MTILE   128
#define THREADS 256
#define HPAD    520   // padded LDS row stride (elements) -> conflict-free ds_load_b128

typedef __attribute__((ext_vector_type(16))) __bf16        v16bf;
typedef __attribute__((ext_vector_type(8)))  float         v8f;
typedef __attribute__((ext_vector_type(8)))  unsigned int  v8u;

struct Pack32 { uint4 lo, hi; };   // 32 bytes -> bit_cast to v16bf

__device__ __forceinline__ unsigned short f2bf(float f) {
  unsigned int u = __builtin_bit_cast(unsigned int, f);
  u += 0x7FFFu + ((u >> 16) & 1u);           // round-to-nearest-even
  return (unsigned short)(u >> 16);
}
__device__ __forceinline__ __bf16 f2bf16(float f) {
  unsigned short s = f2bf(f);
  return __builtin_bit_cast(__bf16, s);
}
__device__ __forceinline__ float fast_sigmoid(float x) {
  return __builtin_amdgcn_rcpf(1.0f + __expf(-x));
}
__device__ __forceinline__ float fast_softplus(float x) {
  return fmaxf(x, 0.0f) + __logf(1.0f + __expf(-fabsf(x)));
}

// A fragment: contiguous k order (16x32): lane run [hi*16,+16)
__device__ __forceinline__ v16bf load_a_frag(const __bf16* row, int kt, int hi) {
  Pack32 p;
  p.lo = *(const uint4*)(row + kt * 32 + hi * 16);
  p.hi = *(const uint4*)(row + kt * 32 + hi * 16 + 8);
  return __builtin_bit_cast(v16bf, p);
}
// B fragment: packed layout -> one contiguous 32B run per lane
__device__ __forceinline__ v16bf load_b_frag(const __bf16* Wp, int kt, int n, int hi) {
  return __builtin_bit_cast(v16bf, *(const v8u*)(Wp + ((kt * HDIM + n) * 32 + hi * 16)));
}

// Async global->LDS copy (ASYNCcnt path). Flat LDS pointer low 32 bits are the
// LDS offset (ISA 10.2), which is what VDST expects.
__device__ __forceinline__ void async_g2l_b32(const float* g, float* l) {
  unsigned lds_off = (unsigned)(uintptr_t)l;
  unsigned long long ga = (unsigned long long)(uintptr_t)g;
  asm volatile("global_load_async_to_lds_b32 %0, %1, off"
               :: "v"(lds_off), "v"(ga) : "memory");
}

// ---------------------------------------------------------------------------
// Prep: pack W2 (512x512 f32, row-major [k][n]) into bf16 B-operand layout for
//   z2 = h1 @ W2    (W2b : tile kt, column n, 32 contiguous k')
//   g1 = v2 @ W2^T  (W2tb: same layout built from W2[n][k])
// ---------------------------------------------------------------------------
__global__ __launch_bounds__(256)
void pack_w2_kernel(const float* __restrict__ W2,
                    __bf16* __restrict__ W2b,
                    __bf16* __restrict__ W2tb) {
  int idx = blockIdx.x * 256 + threadIdx.x;        // 0 .. 512*512-1
  int k = idx >> 9, n = idx & (HDIM - 1);
  int dst = ((k >> 5) * HDIM + n) * 32 + (k & 31);
  W2b [dst] = f2bf16(W2[k * HDIM + n]);
  W2tb[dst] = f2bf16(W2[n * HDIM + k]);
}

// ---------------------------------------------------------------------------
// Fused Port-Hamiltonian gradient kernel: one 128-row batch tile per WGP.
// ---------------------------------------------------------------------------
__global__ __launch_bounds__(THREADS)
void phnet_fused_kernel(const float* __restrict__ state,       // [B,2]
                        const float* __restrict__ action_emb,  // [B,8]
                        const float* __restrict__ W1,          // [2,512]
                        const float* __restrict__ b1,          // [512]
                        const float* __restrict__ b2,          // [512]
                        const float* __restrict__ W3,          // [512,1]
                        const float* __restrict__ log_damping, // [1]
                        const float* __restrict__ Gw,          // [8,1]
                        const float* __restrict__ Gb,          // [1]
                        const __bf16* __restrict__ W2b,
                        const __bf16* __restrict__ W2tb,
                        float* __restrict__ out)               // [B,2]
{
  extern __shared__ char smem[];
  __bf16* lds_h1   = (__bf16*)smem;                              // 128*520*2 = 133120
  __bf16* lds_v2   = (__bf16*)(smem + MTILE * HPAD * 2);         // 133120
  float*  lds_W1   = (float*)(smem + 2 * MTILE * HPAD * 2);      // 1024
  float*  lds_b1   = lds_W1 + 2 * HDIM;                          // 512
  float*  lds_b2   = lds_b1 + HDIM;                              // 512
  float*  lds_w3   = lds_b2 + HDIM;                              // 512
  float*  lds_st   = lds_w3 + HDIM;                              // 256  (state tile)
  float*  lds_dH   = lds_st + 2 * MTILE;                         // 256

  const int tid = threadIdx.x;
  const long base_row = (long)blockIdx.x * MTILE;

  // ---- Phase 0: stage parameters + state tile via async global->LDS -------
  for (int i = tid; i < 2 * HDIM; i += THREADS) async_g2l_b32(W1 + i, lds_W1 + i);
  for (int i = tid; i < HDIM; i += THREADS) {
    async_g2l_b32(b1 + i, lds_b1 + i);
    async_g2l_b32(b2 + i, lds_b2 + i);
    async_g2l_b32(W3 + i, lds_w3 + i);
  }
  async_g2l_b32(state + base_row * 2 + tid, lds_st + tid);  // 256 floats
  lds_dH[tid] = 0.0f;
  asm volatile("s_wait_asynccnt 0x0" ::: "memory");
  __syncthreads();

  // ---- Phase 1: z1 = s@W1+b1 (VALU), h1 = softplus(z1) -> LDS bf16 --------
  for (int idx = tid; idx < MTILE * HDIM; idx += THREADS) {
    int m = idx >> 9, j = idx & (HDIM - 1);
    float z = fmaf(lds_st[2 * m], lds_W1[j],
              fmaf(lds_st[2 * m + 1], lds_W1[HDIM + j], lds_b1[j]));
    lds_h1[m * HPAD + j] = f2bf16(fast_softplus(z));
  }
  __syncthreads();

  const int wave = tid >> 5;        // 0..7 : owns rows [16*wave, 16*wave+16)
  const int lane = tid & 31;
  const int lrow = lane & 15;       // A: row / B,C,D: column-in-tile
  const int hi   = lane >> 4;       // lane-half select per ISA layouts

  // ---- Phase 2: z2 = h1 @ W2 (WMMA bf16), v2 = sigmoid(z2+b2)*w3 ----------
  {
    const __bf16* h1row = lds_h1 + (wave * 16 + lrow) * HPAD;
    for (int ng = 0; ng < 8; ++ng) {
      v8f acc[4] = {};
      v16bf Acur = load_a_frag(h1row, 0, hi);
      v16bf Bcur[4];
#pragma unroll
      for (int i = 0; i < 4; ++i)
        Bcur[i] = load_b_frag(W2b, 0, (ng * 4 + i) * 16 + lrow, hi);
#pragma unroll
      for (int kt = 0; kt < 16; ++kt) {
        v16bf Anx{};
        v16bf Bnx[4];
        if (kt < 15) {                          // software pipeline: k+1 in flight
          Anx = load_a_frag(h1row, kt + 1, hi);
#pragma unroll
          for (int i = 0; i < 4; ++i)
            Bnx[i] = load_b_frag(W2b, kt + 1, (ng * 4 + i) * 16 + lrow, hi);
        }
        if (kt + 4 < 16)                        // WGP-scope prefetch (locality 3)
          __builtin_prefetch(W2b + (((kt + 4) * HDIM + ng * 64 + lrow) * 32), 0, 3);
#pragma unroll
        for (int i = 0; i < 4; ++i)
          acc[i] = __builtin_amdgcn_wmma_f32_16x16x32_bf16(
                     false, Acur, false, Bcur[i], (short)0, acc[i], false, false);
        Acur = Anx;
#pragma unroll
        for (int i = 0; i < 4; ++i) Bcur[i] = Bnx[i];
      }
#pragma unroll
      for (int i = 0; i < 4; ++i) {
        int n = (ng * 4 + i) * 16 + lrow;
        float b2n = lds_b2[n], w3n = lds_w3[n];
#pragma unroll
        for (int r = 0; r < 8; ++r) {           // C layout: row = r + 8*hi
          int row = wave * 16 + hi * 8 + r;
          float v = fast_sigmoid(acc[i][r] + b2n) * w3n;
          lds_v2[row * HPAD + n] = f2bf16(v);
        }
      }
    }
  }
  __syncthreads();

  // ---- Phase 3: g1 = v2 @ W2^T (WMMA); u = sigmoid(z1)*g1; dH += u@W1^T ---
  {
    float p0[8], p1[8];
#pragma unroll
    for (int r = 0; r < 8; ++r) { p0[r] = 0.0f; p1[r] = 0.0f; }
    const __bf16* v2row = lds_v2 + (wave * 16 + lrow) * HPAD;
    for (int ng = 0; ng < 8; ++ng) {
      v8f acc[4] = {};
      v16bf Acur = load_a_frag(v2row, 0, hi);
      v16bf Bcur[4];
#pragma unroll
      for (int i = 0; i < 4; ++i)
        Bcur[i] = load_b_frag(W2tb, 0, (ng * 4 + i) * 16 + lrow, hi);
#pragma unroll
      for (int kt = 0; kt < 16; ++kt) {
        v16bf Anx{};
        v16bf Bnx[4];
        if (kt < 15) {
          Anx = load_a_frag(v2row, kt + 1, hi);
#pragma unroll
          for (int i = 0; i < 4; ++i)
            Bnx[i] = load_b_frag(W2tb, kt + 1, (ng * 4 + i) * 16 + lrow, hi);
        }
        if (kt + 4 < 16)
          __builtin_prefetch(W2tb + (((kt + 4) * HDIM + ng * 64 + lrow) * 32), 0, 3);
#pragma unroll
        for (int i = 0; i < 4; ++i)
          acc[i] = __builtin_amdgcn_wmma_f32_16x16x32_bf16(
                     false, Acur, false, Bcur[i], (short)0, acc[i], false, false);
        Acur = Anx;
#pragma unroll
        for (int i = 0; i < 4; ++i) Bcur[i] = Bnx[i];
      }
#pragma unroll
      for (int i = 0; i < 4; ++i) {
        int n = (ng * 4 + i) * 16 + lrow;
        float w10 = lds_W1[n], w11 = lds_W1[HDIM + n], b1n = lds_b1[n];
#pragma unroll
        for (int r = 0; r < 8; ++r) {
          int row = wave * 16 + hi * 8 + r;
          float z1 = fmaf(lds_st[2 * row], w10,
                     fmaf(lds_st[2 * row + 1], w11, b1n));   // recompute (cheap)
          float u = fast_sigmoid(z1) * acc[i][r];
          p0[r] = fmaf(u, w10, p0[r]);
          p1[r] = fmaf(u, w11, p1[r]);
        }
      }
    }
#pragma unroll
    for (int r = 0; r < 8; ++r) {
      int row = wave * 16 + hi * 8 + r;
      atomicAdd(&lds_dH[2 * row],     p0[r]);   // ds_add_f32
      atomicAdd(&lds_dH[2 * row + 1], p1[r]);
    }
  }
  __syncthreads();

  // ---- Phase 4: dq_dt = dH_dp ; dp_dt = -dH_dq - damping*dH_dp + G_u ------
  if (tid < MTILE) {
    long gm = base_row + tid;
    float dHq = lds_dH[2 * tid], dHp = lds_dH[2 * tid + 1];
    float damping = __expf(log_damping[0]);
    const float* ae = action_emb + gm * 8;
    float gu = Gb[0];
#pragma unroll
    for (int e = 0; e < 8; ++e) gu = fmaf(ae[e], Gw[e], gu);
    out[gm * 2 + 0] = dHp;
    out[gm * 2 + 1] = -dHq - damping * dHp + gu;
  }
}

// ---------------------------------------------------------------------------
extern "C" void kernel_launch(void* const* d_in, const int* in_sizes, int n_in,
                              void* d_out, int out_size, void* d_ws, size_t ws_size,
                              hipStream_t stream) {
  // setup_inputs order: t, state, action_emb, W1, b1, W2, b2, W3, b3,
  //                     log_damping, Gw, Gb
  const float* state       = (const float*)d_in[1];
  const float* action_emb  = (const float*)d_in[2];
  const float* W1          = (const float*)d_in[3];
  const float* b1          = (const float*)d_in[4];
  const float* W2          = (const float*)d_in[5];
  const float* b2          = (const float*)d_in[6];
  const float* W3          = (const float*)d_in[7];
  const float* log_damping = (const float*)d_in[9];
  const float* Gw          = (const float*)d_in[10];
  const float* Gb          = (const float*)d_in[11];
  float* out = (float*)d_out;

  __bf16* W2b  = (__bf16*)d_ws;                 // 512 KB
  __bf16* W2tb = W2b + HDIM * HDIM;             // 512 KB

  pack_w2_kernel<<<(HDIM * HDIM) / 256, 256, 0, stream>>>(W2, W2b, W2tb);

  constexpr size_t kSmem = 2 * MTILE * HPAD * 2                       // h1 + v2 (bf16)
                         + (2 * HDIM + 3 * HDIM + 2 * MTILE + 2 * MTILE) * sizeof(float);
  hipFuncSetAttribute(reinterpret_cast<const void*>(&phnet_fused_kernel),
                      hipFuncAttributeMaxDynamicSharedMemorySize, (int)kSmem);

  phnet_fused_kernel<<<BATCH / MTILE, THREADS, kSmem, stream>>>(
      state, action_emb, W1, b1, b2, W3, log_damping, Gw, Gb, W2b, W2tb, out);
}
